// Attention_11570641896162
// MI455X (gfx1250) — matrix operand order repaired and checked by reading, not compiled, not run on hardware
//
#include <hip/hip_runtime.h>

typedef __attribute__((ext_vector_type(16))) __bf16 v16bf;
typedef __attribute__((ext_vector_type(8)))  __bf16 v8bf;
typedef __attribute__((ext_vector_type(8)))  float  v8f;

#define EPSV 1e-5f

__device__ __forceinline__ float fsilu(float v){ return v / (1.0f + __expf(-v)); }
__device__ __forceinline__ __bf16 f2bf(float f){ return (__bf16)f; }

// ---- WMMA operand loaders (wave32, ISA 7.12.2 layouts) -----------------
// A operand: 16x32 bf16 tile, row-major source, leading dim ld.
// lane L holds row M=L&15; K elems {kb..kb+7} and {kb+16..kb+23}, kb=(L>=16)*8.
__device__ __forceinline__ v16bf load_a(const __bf16* src, int ld){
  int lane = threadIdx.x & 31;
  int row  = lane & 15;
  int kb   = (lane >> 4) << 3;            // 0 or 8
  const __bf16* p = src + row * ld + kb;
  v8bf lo = *reinterpret_cast<const v8bf*>(p);
  v8bf hi = *reinterpret_cast<const v8bf*>(p + 16);
  v16bf a;
#pragma unroll
  for (int i = 0; i < 8; ++i){ a[i] = lo[i]; a[i+8] = hi[i]; }
  return a;
}
// B operand: 32x16 bf16 tile, loaded from *transposed* storage where rows are
// the N dimension and the K dim is contiguous. lane L holds N=L&15,
// K elems {kb..kb+7} and {kb+16..kb+23}, kb=(L>=16)*8.
__device__ __forceinline__ v16bf load_b(const __bf16* src, int ld){
  int lane = threadIdx.x & 31;
  int row  = lane & 15;
  int kb   = (lane >> 4) << 3;            // 0 or 8
  const __bf16* p = src + row * ld + kb;
  v8bf lo = *reinterpret_cast<const v8bf*>(p);
  v8bf hi = *reinterpret_cast<const v8bf*>(p + 16);
  v16bf b;
#pragma unroll
  for (int i = 0; i < 8; ++i){ b[i] = lo[i]; b[i+8] = hi[i]; }
  return b;
}

// ---- 0) weight prep: fold BN scale, convert to bf16, biases ------------
__global__ void k_prep(const float* qkv_w, const float* qg, const float* qb,
                       const float* qm, const float* qv,
                       const float* proj_w, const float* pg, const float* pb,
                       const float* pm, const float* pv,
                       __bf16* wqkv, __bf16* wp, float* bq, float* bp)
{
  int t = blockIdx.x * blockDim.x + threadIdx.x;
  if (t < 512*256){
    int o = t >> 8;
    float inv = qg[o] * rsqrtf(qv[o] + EPSV);
    wqkv[t] = f2bf(qkv_w[t] * inv);
  } else if (t < 512*256 + 256*256){
    int u = t - 512*256;
    int o = u >> 8;
    float inv = pg[o] * rsqrtf(pv[o] + EPSV);
    wp[u] = f2bf(proj_w[u] * inv);
  } else if (t < 512*256 + 256*256 + 512){
    int o = t - (512*256 + 256*256);
    float inv = qg[o] * rsqrtf(qv[o] + EPSV);
    bq[o] = qb[o] - qm[o] * inv;
  } else if (t < 512*256 + 256*256 + 512 + 256){
    int o = t - (512*256 + 256*256 + 512);
    float inv = pg[o] * rsqrtf(pv[o] + EPSV);
    bp[o] = pb[o] - pm[o] * inv;
  }
}

// ---- 1) depthwise 3x3 + BN, v = x + pe; also x -> bf16 (N,C) -----------
__global__ void k_pe_v(const float* x, const float* pe_w, const float* g,
                       const float* be, const float* mn, const float* vr,
                       __bf16* vbf, __bf16* xb)
{
  int t  = blockIdx.x * blockDim.x + threadIdx.x;   // [0, 4*256*1600)
  int hw = t % 1600;
  int bc = t / 1600;
  int c  = bc & 255;
  int b  = bc >> 8;
  int yy = hw / 40, xx = hw % 40;
  const float* xs = x + bc * 1600;
  float acc = 0.f;
#pragma unroll
  for (int dy = -1; dy <= 1; ++dy){
    int y2 = yy + dy;
    if ((unsigned)y2 >= 40u) continue;
#pragma unroll
    for (int dx = -1; dx <= 1; ++dx){
      int x2 = xx + dx;
      if ((unsigned)x2 >= 40u) continue;
      acc += xs[y2*40 + x2] * pe_w[c*9 + (dy+1)*3 + (dx+1)];
    }
  }
  float inv = g[c] * rsqrtf(vr[c] + EPSV);
  float pe  = acc * inv + (be[c] - mn[c] * inv);
  float xv  = xs[hw];
  vbf[t] = f2bf(xv + pe);                           // (b*256+c)*1600 + hw
  xb[(b*1600 + hw)*256 + c] = f2bf(xv);             // transposed for qkv GEMM
}

// ---- 2) qkv 1x1 GEMM: [512x256]x[256x6400], fused bias+SiLU ------------
__global__ __launch_bounds__(256) void k_qkv(const __bf16* wqkv, const __bf16* xb,
                                             const float* bq, __bf16* qbuf, __bf16* kbuf)
{
  int wave = threadIdx.x >> 5;
  int lane = threadIdx.x & 31;
  int tile = blockIdx.x * 8 + wave;   // [0, 32*400)
  int ot = tile / 400, nt = tile % 400;
  const __bf16* aBase = wqkv + ot*16*256;
  const __bf16* bBase = xb   + nt*16*256;
  v8f c = {};
#pragma unroll
  for (int kk = 0; kk < 256; kk += 32){
    if (kk < 224){ __builtin_prefetch(aBase + kk + 32); __builtin_prefetch(bBase + kk + 32); }
    v16bf a = load_a(aBase + kk, 256);
    v16bf b = load_b(bBase + kk, 256);
    c = __builtin_amdgcn_wmma_f32_16x16x32_bf16(false, a, false, b, (short)0, c, false, false);
  }
  int half = lane >> 4, col = lane & 15;
  int n = nt*16 + col;
  int bi = n / 1600, nl = n % 1600;
#pragma unroll
  for (int r = 0; r < 8; ++r){
    int o = ot*16 + r + half*8;
    float v = fsilu(c[r] + bq[o]);
    if (o < 256){
      int h = o >> 5, d = o & 31;
      qbuf[((bi*8 + h)*1600 + nl)*32 + d] = f2bf(v);
    } else {
      int cc = o - 256;
      int h = cc >> 5, d = cc & 31;
      kbuf[((bi*8 + h)*1600 + nl)*32 + d] = f2bf(v);
    }
  }
}

// ---- 3) flash attention: per (b,h,16-query tile) wave ------------------
__global__ __launch_bounds__(128) void k_attn(const __bf16* qbuf, const __bf16* kbuf,
                                              const __bf16* vbf, __bf16* obuf)
{
  __shared__ __align__(16) __bf16 sP[4][16*32];
  int wave = threadIdx.x >> 5;
  int lane = threadIdx.x & 31;
  int tile = blockIdx.x * 4 + wave;   // [0, 32*100)
  int bh = tile / 100, nt = tile % 100;
  int b = bh >> 3, h = bh & 7;
  int half = lane >> 4, col = lane & 15;
  __bf16* sp = sP[wave];

  v16bf qa = load_a(qbuf + (bh*1600 + nt*16)*32, 32);   // Q tile, full K=d=32
  float mrun[8], lrun[8];
#pragma unroll
  for (int r = 0; r < 8; ++r){ mrun[r] = -1e30f; lrun[r] = 0.f; }
  v8f acc0 = {}, acc1 = {}, z = {};
  const float scale = 0.17677669529663687f;             // 32^-0.5

  for (int mp = 0; mp < 50; ++mp){
    const __bf16* K0 = kbuf + (bh*1600 + mp*32)*32;
    v16bf kb0 = load_b(K0, 32);
    v16bf kb1 = load_b(K0 + 16*32, 32);
    v8f s0 = __builtin_amdgcn_wmma_f32_16x16x32_bf16(false, qa, false, kb0, (short)0, z, false, false);
    v8f s1 = __builtin_amdgcn_wmma_f32_16x16x32_bf16(false, qa, false, kb1, (short)0, z, false, false);
#pragma unroll
    for (int r = 0; r < 8; ++r){
      float x0 = s0[r]*scale, x1 = s1[r]*scale;
      float mx = fmaxf(x0, x1);                         // row = fixed within half-wave
      mx = fmaxf(mx, __shfl_xor(mx, 1));
      mx = fmaxf(mx, __shfl_xor(mx, 2));
      mx = fmaxf(mx, __shfl_xor(mx, 4));
      mx = fmaxf(mx, __shfl_xor(mx, 8));
      float mnew = fmaxf(mrun[r], mx);
      float corr = __expf(mrun[r] - mnew);
      mrun[r] = mnew;
      float p0 = __expf(x0 - mnew), p1 = __expf(x1 - mnew);
      float rs = p0 + p1;
      rs += __shfl_xor(rs, 1); rs += __shfl_xor(rs, 2);
      rs += __shfl_xor(rs, 4); rs += __shfl_xor(rs, 8);
      lrun[r] = lrun[r]*corr + rs;
      acc0[r] *= corr; acc1[r] *= corr;
      int n = r + half*8;                               // C-layout row of score tile
      sp[n*32 + col]      = f2bf(p0);
      sp[n*32 + col + 16] = f2bf(p1);
    }
    asm volatile("s_wait_dscnt 0x0" ::: "memory");      // LDS stores -> loads fence
    v16bf pa = load_a(sp, 32);                          // P tile in A layout (16n x 32m)
    const __bf16* V0 = vbf + (bh*32)*1600 + mp*32;      // rows = d, ld = 1600
    v16bf vb0 = load_b(V0, 1600);                       // d = 0..15
    v16bf vb1 = load_b(V0 + 16*1600, 1600);             // d = 16..31
    acc0 = __builtin_amdgcn_wmma_f32_16x16x32_bf16(false, pa, false, vb0, (short)0, acc0, false, false);
    acc1 = __builtin_amdgcn_wmma_f32_16x16x32_bf16(false, pa, false, vb1, (short)0, acc1, false, false);
  }
#pragma unroll
  for (int r = 0; r < 8; ++r){
    int nglob = nt*16 + r + half*8;
    float invl = 1.0f / lrun[r];
    __bf16* dst = obuf + (b*1600 + nglob)*256 + h*32;   // (N,C) transposed for proj
    dst[col]      = f2bf(acc0[r]*invl);
    dst[col + 16] = f2bf(acc1[r]*invl);
  }
}

// ---- 4) proj 1x1 GEMM: [256x256]x[256x6400], fused bias+SiLU, fp32 out -
__global__ __launch_bounds__(256) void k_proj(const __bf16* wp, const __bf16* obuf,
                                              const float* bp, float* out)
{
  int wave = threadIdx.x >> 5;
  int lane = threadIdx.x & 31;
  int tile = blockIdx.x * 8 + wave;   // [0, 16*400)
  int ot = tile / 400, nt = tile % 400;
  const __bf16* aBase = wp   + ot*16*256;
  const __bf16* bBase = obuf + nt*16*256;
  v8f c = {};
#pragma unroll
  for (int kk = 0; kk < 256; kk += 32){
    if (kk < 224){ __builtin_prefetch(aBase + kk + 32); __builtin_prefetch(bBase + kk + 32); }
    v16bf a = load_a(aBase + kk, 256);
    v16bf b = load_b(bBase + kk, 256);
    c = __builtin_amdgcn_wmma_f32_16x16x32_bf16(false, a, false, b, (short)0, c, false, false);
  }
  int half = lane >> 4, col = lane & 15;
  int n = nt*16 + col;
  int bi = n / 1600, nl = n % 1600;
#pragma unroll
  for (int r = 0; r < 8; ++r){
    int o = ot*16 + r + half*8;
    out[(bi*256 + o)*1600 + nl] = fsilu(c[r] + bp[o]);
  }
}

extern "C" void kernel_launch(void* const* d_in, const int* in_sizes, int n_in,
                              void* d_out, int out_size, void* d_ws, size_t ws_size,
                              hipStream_t stream) {
  const float* x     = (const float*)d_in[0];
  const float* qkv_w = (const float*)d_in[1];
  const float* qg    = (const float*)d_in[2];
  const float* qb    = (const float*)d_in[3];
  const float* qm    = (const float*)d_in[4];
  const float* qv    = (const float*)d_in[5];
  const float* pw    = (const float*)d_in[6];
  const float* pg    = (const float*)d_in[7];
  const float* pb    = (const float*)d_in[8];
  const float* pm    = (const float*)d_in[9];
  const float* pv    = (const float*)d_in[10];
  const float* pe_w  = (const float*)d_in[11];
  const float* peg   = (const float*)d_in[12];
  const float* peb   = (const float*)d_in[13];
  const float* pem   = (const float*)d_in[14];
  const float* pev   = (const float*)d_in[15];
  float* out = (float*)d_out;

  char* ws = (char*)d_ws;
  __bf16* xb   = (__bf16*)(ws + 0);          // 6400*256 bf16 = 3,276,800 B
  __bf16* vbf  = (__bf16*)(ws + 3276800);
  __bf16* qbuf = (__bf16*)(ws + 6553600);
  __bf16* kbuf = (__bf16*)(ws + 9830400);
  __bf16* obuf = (__bf16*)(ws + 13107200);
  __bf16* wqkv = (__bf16*)(ws + 16384000);   // 512*256 bf16
  __bf16* wpb  = (__bf16*)(ws + 16646144);   // 256*256 bf16
  float*  bq   = (float*) (ws + 16777216);   // 512 f32
  float*  bpv  = (float*) (ws + 16779264);   // 256 f32

  k_prep<<<771, 256, 0, stream>>>(qkv_w, qg, qb, qm, qv, pw, pg, pb, pm, pv,
                                  wqkv, wpb, bq, bpv);
  k_pe_v<<<6400, 256, 0, stream>>>(x, pe_w, peg, peb, pem, pev, vbf, xb);
  k_qkv<<<1600, 256, 0, stream>>>(wqkv, xb, bq, qbuf, kbuf);
  k_attn<<<800, 128, 0, stream>>>(qbuf, kbuf, vbf, obuf);
  k_proj<<<800, 256, 0, stream>>>(wpb, obuf, bpv, out);
}
